// FlowWarp_60576218743324
// MI455X (gfx1250) — matrix operand order repaired and checked by reading, not compiled, not run on hardware
//
#include <hip/hip_runtime.h>
#include <stdint.h>

// Problem constants
#define B_   8
#define H_   64
#define W_   64
#define C_   512
#define M_   (B_ * H_ * W_)   // 32768 rows
#define KC_  1026             // real K (512 + 512 + 2)
#define KP_  1056             // K padded to multiple of 32 (33 tiles)
#define N_   1024
#define KTILES (KP_ / 32)     // 33

typedef __attribute__((ext_vector_type(16))) __bf16 v16bf;
typedef __attribute__((ext_vector_type(8)))  float  v8f;
typedef __attribute__((ext_vector_type(4)))  int    v4i;
typedef __attribute__((address_space(1))) v4i gv4i;   // global v4i
typedef __attribute__((address_space(3))) v4i lv4i;   // LDS v4i

union Frag16 { uint4 q[2]; v16bf v; };

__device__ __forceinline__ unsigned short f2bf(float f) {
    union { float f; uint32_t u; } x;
    x.f = f;
    uint32_t u = x.u;
    u += 0x7FFFu + ((u >> 16) & 1u);   // round-to-nearest-even
    return (unsigned short)(u >> 16);
}

// CDNA5 async global->LDS (ASYNCcnt path), builtin with inline-asm fallback.
__device__ __forceinline__ void async_b128(const unsigned short* g, unsigned short* l) {
#if __has_builtin(__builtin_amdgcn_global_load_async_to_lds_b128)
    // Param types: (v4i addrspace(1)*, v4i addrspace(3)*, imm offset, imm cpol).
    // Go through integers: addrspace(3) pointers are 32-bit (= low 32 bits of
    // the generic address on AMDGPU), addrspace(1) pointers are 64-bit.
    gv4i* gp = (gv4i*)(uintptr_t)g;
    lv4i* lp = (lv4i*)(uint32_t)(uintptr_t)l;
    __builtin_amdgcn_global_load_async_to_lds_b128(gp, lp, 0, 0);
#else
    asm volatile("global_load_async_to_lds_b128 %0, %1, off"
                 :: "v"((uint32_t)(uintptr_t)l), "v"(g) : "memory");
#endif
}

__device__ __forceinline__ void wait_async0() {
#if __has_builtin(__builtin_amdgcn_s_wait_asynccnt)
    __builtin_amdgcn_s_wait_asynccnt(0);
#else
    asm volatile("s_wait_asynccnt 0" ::: "memory");
#endif
}

// ---------------------------------------------------------------------------
// Stage 1: bilinear warp + concat -> A[M][KP_] in bf16 (zero-padded K tail).
// One wave32 per pixel; flow is wave-uniform, channels vectorized over lanes.
// ---------------------------------------------------------------------------
__global__ __launch_bounds__(256) void prep_a_kernel(
    const float* __restrict__ x, const float* __restrict__ xp,
    const float* __restrict__ flow, unsigned short* __restrict__ A)
{
    const int wave = threadIdx.x >> 5;
    const int lane = threadIdx.x & 31;
    const int p = blockIdx.x * 8 + wave;       // pixel index == GEMM row m
    const int b  = p >> 12;
    const int yy = (p >> 6) & 63;
    const int xx = p & 63;

    const float fy = flow[2 * p + 0];
    const float fx = flow[2 * p + 1];
    // dense_image_warp(x_prev, -bw_flow): sample at grid + bw_flow
    const float qy = (float)yy + fy;
    const float qx = (float)xx + fx;
    const float fly = fminf(fmaxf(floorf(qy), 0.f), (float)(H_ - 2));
    const float flx = fminf(fmaxf(floorf(qx), 0.f), (float)(W_ - 2));
    const float ay  = fminf(fmaxf(qy - fly, 0.f), 1.f);
    const float ax  = fminf(fmaxf(qx - flx, 0.f), 1.f);
    const int iy = (int)fly, ix = (int)flx;

    const size_t base = (((size_t)b * H_ + iy) * W_ + ix) * C_;
    const float4* tl = (const float4*)(xp + base);
    const float4* tr = (const float4*)(xp + base + C_);
    const float4* bl = (const float4*)(xp + base + (size_t)W_ * C_);
    const float4* br = (const float4*)(xp + base + (size_t)W_ * C_ + C_);
    const float4* xr = (const float4*)(x + (size_t)p * C_);
    unsigned short* arow = A + (size_t)p * KP_;

#pragma unroll
    for (int j = 0; j < 4; ++j) {
        const int q = lane + j * 32;           // float4 group 0..127 (coalesced)
        const float4 a4 = tl[q], b4 = tr[q], c4 = bl[q], d4 = br[q];
        float4 t, bo, w;
        t.x = a4.x + ax * (b4.x - a4.x);  t.y = a4.y + ax * (b4.y - a4.y);
        t.z = a4.z + ax * (b4.z - a4.z);  t.w = a4.w + ax * (b4.w - a4.w);
        bo.x = c4.x + ax * (d4.x - c4.x); bo.y = c4.y + ax * (d4.y - c4.y);
        bo.z = c4.z + ax * (d4.z - c4.z); bo.w = c4.w + ax * (d4.w - c4.w);
        w.x = t.x + ay * (bo.x - t.x);    w.y = t.y + ay * (bo.y - t.y);
        w.z = t.z + ay * (bo.z - t.z);    w.w = t.w + ay * (bo.w - t.w);

        ushort4 ws, xs;
        ws.x = f2bf(w.x); ws.y = f2bf(w.y); ws.z = f2bf(w.z); ws.w = f2bf(w.w);
        const float4 xv = xr[q];
        xs.x = f2bf(xv.x); xs.y = f2bf(xv.y); xs.z = f2bf(xv.z); xs.w = f2bf(xv.w);

        *(ushort4*)(arow + q * 4)        = xs;   // channels [0,512)
        *(ushort4*)(arow + C_ + q * 4)   = ws;   // channels [512,1024)
    }
    if (lane == 0) {
        arow[1024] = f2bf(fy);                   // flow channel 0
        arow[1025] = f2bf(fx);                   // flow channel 1
    } else if (lane >= 2) {
        arow[1024 + lane] = 0;                   // zero K padding 1026..1055
    }
}

// ---------------------------------------------------------------------------
// Stage 2: B^T[n][k] = bf16(pw[k][n] * dw_kernel[k]); depthwise folded in.
// ---------------------------------------------------------------------------
__global__ __launch_bounds__(256) void prep_b_kernel(
    const float* __restrict__ pw, const float* __restrict__ dwk,
    unsigned short* __restrict__ BT)
{
    const int n = blockIdx.x;
    for (int k = threadIdx.x; k < KP_; k += 256) {
        const float v = (k < KC_) ? pw[(size_t)k * N_ + n] * dwk[k] : 0.f;
        BT[(size_t)n * KP_ + k] = f2bf(v);
    }
}

// ---------------------------------------------------------------------------
// Stage 3: GEMM via v_wmma_f32_16x16x32_bf16.
// Block: 256 thr = 8 waves (2x4), tile 128x256. Wave: 64x64 = 4x4 accums.
// Global->LDS via async b128 (ASYNCcnt) with double-buffered LDS tiles.
// ---------------------------------------------------------------------------
__global__ __launch_bounds__(256) void gemm_kernel(
    const unsigned short* __restrict__ A, const unsigned short* __restrict__ BT,
    const float* __restrict__ bias, float* __restrict__ out)
{
    __shared__ unsigned short sA[2][128 * 32];   // 2 x 8 KB   (M x K tile)
    __shared__ unsigned short sB[2][256 * 32];   // 2 x 16 KB  (N x K tile, K contig)

    const int tileN = (blockIdx.x & 3) * 256;
    const int tileM = (blockIdx.x >> 2) * 128;
    const int tid   = threadIdx.x;
    const int wave  = tid >> 5;
    const int lane  = tid & 31;
    const int waveM = (wave >> 2) * 64;          // 0 / 64
    const int waveN = (wave & 3) * 64;           // 0 / 64 / 128 / 192
    const int l15   = lane & 15;
    const int kgA   = (lane >> 4) * 16;          // A: K 0-15 / 16-31 halves (matches B)
    const int kgB   = (lane >> 4) * 16;          // B: K 0-15 / 16-31 halves (contiguous, matches A)

    v8f acc[4][4];
#pragma unroll
    for (int i = 0; i < 4; ++i)
#pragma unroll
        for (int j = 0; j < 4; ++j)
            acc[i][j] = (v8f){0.f, 0.f, 0.f, 0.f, 0.f, 0.f, 0.f, 0.f};

    // Cooperative async staging:
    //  A tile: 128 rows x 64B -> 2 threads/row, 2x b128 each.
    //  B tile: 256 rows x 64B -> 1 thread/row,  4x b128 each.
    const int rowA = tid >> 1;
    const int offA = (tid & 1) * 16;             // ushort units (32B half-row)
    const unsigned short* gA = A  + (size_t)(tileM + rowA) * KP_ + offA;
    const unsigned short* gB = BT + (size_t)(tileN + tid) * KP_;

    auto load_tiles = [&](int kt, int buf) {
        const unsigned short* ga = gA + kt * 32;
        const unsigned short* gb = gB + kt * 32;
        unsigned short* da = &sA[buf][rowA * 32 + offA];
        unsigned short* db = &sB[buf][tid * 32];
        async_b128(ga,      da);
        async_b128(ga + 8,  da + 8);
        async_b128(gb,      db);
        async_b128(gb + 8,  db + 8);
        async_b128(gb + 16, db + 16);
        async_b128(gb + 24, db + 24);
    };

    load_tiles(0, 0);
    wait_async0();
    __syncthreads();

    for (int kt = 0; kt < KTILES; ++kt) {
        const int cur = kt & 1;
        if (kt + 1 < KTILES) {
            load_tiles(kt + 1, cur ^ 1);         // overlap with compute below
            __builtin_prefetch(gA + (kt + 1) * 32, 0, 1);  // global_prefetch_b8
            __builtin_prefetch(gB + (kt + 1) * 32, 0, 1);
        }

        // B fragments: lane = column N, 16 contiguous K per lane (2x ds_load_b128)
        Frag16 bf[4];
#pragma unroll
        for (int j = 0; j < 4; ++j) {
            const uint4* pb = (const uint4*)(&sB[cur][(waveN + j * 16 + l15) * 32 + kgB]);
            bf[j].q[0] = pb[0];
            bf[j].q[1] = pb[1];
        }

#pragma unroll
        for (int i = 0; i < 4; ++i) {
            // A fragment: lane = row M; elems 0-15 = K kg..kg+15 (contiguous)
            const unsigned short* pa = &sA[cur][(waveM + i * 16 + l15) * 32 + kgA];
            Frag16 af;
            af.q[0] = *(const uint4*)(pa);
            af.q[1] = *(const uint4*)(pa + 8);
#pragma unroll
            for (int j = 0; j < 4; ++j)
                acc[i][j] = __builtin_amdgcn_wmma_f32_16x16x32_bf16(
                    false, af.v, false, bf[j].v, (short)0, acc[i][j], false, false);
        }

        wait_async0();      // next buffer fully written (this wave's asyncs)
        __syncthreads();    // all waves done reading cur + done waiting
    }

    // Epilogue: C/D layout — VGPR r: lanes 0-15 -> M=r, lanes 16-31 -> M=r+8.
    const int rHalf = (lane >> 4) * 8;
#pragma unroll
    for (int i = 0; i < 4; ++i) {
#pragma unroll
        for (int j = 0; j < 4; ++j) {
            const int gn = tileN + waveN + j * 16 + l15;
            const float bv = bias[gn];
#pragma unroll
            for (int r = 0; r < 8; ++r) {
                const int gm = tileM + waveM + i * 16 + rHalf + r;
                out[(size_t)gm * N_ + gn] = acc[i][j][r] + bv;
            }
        }
    }
}

// ---------------------------------------------------------------------------
extern "C" void kernel_launch(void* const* d_in, const int* in_sizes, int n_in,
                              void* d_out, int out_size, void* d_ws, size_t ws_size,
                              hipStream_t stream) {
    const float* x    = (const float*)d_in[0];
    const float* xp   = (const float*)d_in[1];
    const float* flow = (const float*)d_in[2];
    const float* dwk  = (const float*)d_in[3];
    const float* pw   = (const float*)d_in[4];
    const float* bias = (const float*)d_in[5];
    float* out = (float*)d_out;

    unsigned short* Abuf = (unsigned short*)d_ws;                 // 32768*1056*2 = 69.2 MB
    unsigned short* Bbuf = Abuf + (size_t)M_ * KP_;               // 1024*1056*2  = 2.2 MB

    prep_a_kernel<<<M_ / 8, 256, 0, stream>>>(x, xp, flow, Abuf);
    prep_b_kernel<<<N_, 256, 0, stream>>>(pw, dwk, Bbuf);
    gemm_kernel<<<(M_ / 128) * (N_ / 256), 256, 0, stream>>>(Abuf, Bbuf, bias, out);
}
